// LSTM_Encoder_32023276159140
// MI455X (gfx1250) — compile-verified, matrix operand order checked by reading, not run on hardware
//
#include <hip/hip_runtime.h>

#define DEVI __device__ __forceinline__

typedef __attribute__((ext_vector_type(16))) __bf16 v16bf;
typedef __attribute__((ext_vector_type(8)))  __bf16 bf16x8;
typedef __attribute__((ext_vector_type(4)))  __bf16 bf16x4;
typedef __attribute__((ext_vector_type(8)))  float  v8f;
typedef __attribute__((ext_vector_type(4)))  unsigned u32x4;
typedef __attribute__((ext_vector_type(8)))  unsigned u32x8;

constexpr int B  = 64;     // batch
constexpr int S  = 128;    // seq
constexpr int E  = 1024;   // embed
constexpr int H  = 1024;   // hidden
constexpr int FG = 4 * H;  // 4096 gate width

// ---- workspace layout (all offsets 256B-aligned by construction) ----
constexpr size_t XBF_OFF  = 0;                                    // [S*B, E] bf16   16 MB
constexpr size_t WIH_OFF  = XBF_OFF + (size_t)S * B * E * 2;      // [FG, E] bf16     8 MB
constexpr size_t WHH_OFF  = WIH_OFF + (size_t)FG * E * 2;         // [FG, H] bf16     8 MB
constexpr size_t HBF_OFF  = WHH_OFF + (size_t)FG * H * 2;         // 2x [B, H] bf16 256 KB
constexpr size_t BIAS_OFF = HBF_OFF + 2 * (size_t)B * H * 2;      // [FG] f32        16 KB
constexpr size_t P_OFF    = BIAS_OFF + (size_t)FG * 4;            // [S*B, FG] f32  128 MB
constexpr size_t P_BYTES  = (size_t)S * B * FG * 4;

// ---- bf16 WMMA tile loader: A/B operand, row-major, stride 1024 elems ----
// ISA 7.12.2 16-bit A 16x32 layout: lane holds row (lane&15); K-chunks
// [half*8, half*8+8) and [16+half*8, 16+half*8+8), half = lane>>4.
DEVI v16bf load_tile(const __bf16* p0, int lane) {
  const __bf16* p = p0 + (size_t)(lane & 15) * 1024 + (lane >> 4) * 8;
  bf16x8 lo = *(const bf16x8*)(p);
  bf16x8 hi = *(const bf16x8*)(p + 16);
  return __builtin_shufflevector(lo, hi, 0, 1, 2, 3, 4, 5, 6, 7,
                                         8, 9, 10, 11, 12, 13, 14, 15);
}

DEVI float sigmoidf_(float x) { return 1.0f / (1.0f + __expf(-x)); }

DEVI unsigned lds_offset_of(const void* p) {
  return (unsigned)(unsigned long long)
      (__attribute__((address_space(3))) const char*)p;
}

// ---- zero h, c (in d_out) and both h-bf16 buffers ----
__global__ void init_state(float* out, __bf16* hbf) {
  int i = blockIdx.x * blockDim.x + threadIdx.x;  // 65536 threads
  out[i] = 0.0f;
  out[(size_t)B * H + i] = 0.0f;
  hbf[i] = (__bf16)0.0f;
  hbf[(size_t)B * H + i] = (__bf16)0.0f;
}

// ---- embedding gather + f32->bf16, time-major [s, b, e] ----
__global__ void embed_bf16(const float* __restrict__ emb,
                           const int* __restrict__ inp,
                           __bf16* __restrict__ xbf) {
  int r = blockIdx.x;            // r = s*B + b
  int s = r >> 6;                // / B
  int b = r & (B - 1);
  int tok = inp[b * S + s];      // inputs is [B, S]
  int e = threadIdx.x * 4;
  float4 v = *(const float4*)(emb + (size_t)tok * E + e);
  bf16x4 o;
  o[0] = (__bf16)v.x; o[1] = (__bf16)v.y; o[2] = (__bf16)v.z; o[3] = (__bf16)v.w;
  *(bf16x4*)(xbf + (size_t)r * E + e) = o;
}

// ---- f32 -> bf16 bulk convert (weights) ----
__global__ void cvt_f32_bf16(const float* __restrict__ src,
                             __bf16* __restrict__ dst, int n4) {
  int i = blockIdx.x * blockDim.x + threadIdx.x;
  if (i >= n4) return;
  float4 v = *(const float4*)(src + (size_t)i * 4);
  bf16x4 o;
  o[0] = (__bf16)v.x; o[1] = (__bf16)v.y; o[2] = (__bf16)v.z; o[3] = (__bf16)v.w;
  *(bf16x4*)(dst + (size_t)i * 4) = o;
}

__global__ void bias_sum(const float* __restrict__ bih,
                         const float* __restrict__ bhh,
                         float* __restrict__ bias) {
  int i = blockIdx.x * blockDim.x + threadIdx.x;
  if (i < FG) bias[i] = bih[i] + bhh[i];
}

// ---- big hoisted GEMM: P = Xbf @ Wih^T + bias   [8192 x 4096] ----
__global__ __launch_bounds__(256) void pregate_gemm(
    const __bf16* __restrict__ xbf, const __bf16* __restrict__ wih,
    const float* __restrict__ bias, float* __restrict__ P) {
  int wid  = blockIdx.x * 8 + (threadIdx.x >> 5);  // global wave id
  int lane = threadIdx.x & 31;
  int mt = wid >> 8;    // 0..511  (M tiles over 8192 rows)
  int nt = wid & 255;   // 0..255  (N tiles over 4096 cols)
  const __bf16* arow = xbf + (size_t)(mt * 16) * E;
  const __bf16* brow = wih + (size_t)(nt * 16) * E;  // W_ih rows = B^T cols
  v8f acc = {0.f, 0.f, 0.f, 0.f, 0.f, 0.f, 0.f, 0.f};
  for (int k0 = 0; k0 < E; k0 += 32) {
    v16bf a = load_tile(arow + k0, lane);
    v16bf b = load_tile(brow + k0, lane);
    acc = __builtin_amdgcn_wmma_f32_16x16x32_bf16(false, a, false, b,
                                                  (short)0, acc, false, false);
  }
  int n     = nt * 16 + (lane & 15);
  int mbase = mt * 16 + (lane >> 4) * 8;
  float bv = bias[n];
#pragma unroll
  for (int v = 0; v < 8; ++v)
    P[(size_t)(mbase + v) * FG + n] = acc[v] + bv;
}

// ==== recurrent step, TDM + LDS weight-stationary version ====
// Block covers ALL 4 batch tiles x 2 hidden tiles: its W_hh working set is
// [4 gates][32 rows][1024 K] bf16 = 256 KB, loaded ONCE per block by a single
// 3D Tensor-Data-Mover descriptor (tile 1024 x 32 x 4, plane stride H*1024),
// then the K-loop feeds WMMA's B operand from LDS.
__global__ __launch_bounds__(256) void lstm_step_lds(
    const __bf16* __restrict__ hbf_r, __bf16* __restrict__ hbf_w,
    const __bf16* __restrict__ whh, const float* __restrict__ P,
    float* __restrict__ out, int t) {
  extern __shared__ __bf16 lds[];                  // [4][32][1024] = 256 KB
  int w    = threadIdx.x >> 5;                     // wave 0..7
  int lane = threadIdx.x & 31;
  int mt = w & 3;                                  // batch tile 0..3
  int h2 = w >> 2;                                 // 0..1
  int ht = blockIdx.x * 2 + h2;                    // hidden tile 0..63

  // ---- stage W_hh slice into LDS via one TDM descriptor (wave 0 only) ----
  if (w == 0) {
    unsigned long long ga =
        (unsigned long long)(const void*)(whh + (size_t)(blockIdx.x * 32) * H);
    unsigned lbase = lds_offset_of(lds);
    u32x4 g0;
    g0[0] = 1u;                                    // count=1, user descriptor
    g0[1] = lbase;                                 // lds_addr (bytes)
    g0[2] = (unsigned)(ga & 0xffffffffu);          // global_addr[31:0]
    g0[3] = (unsigned)((ga >> 32) & 0x01ffffffu)   // global_addr[56:32]
            | 0x80000000u;                         // type=2 ("image")
    u32x8 g1;
    g1[0] = 0x00010000u;                           // data_size=1 (2B), no mask
    g1[1] = (unsigned)(1024u << 16);               // tensor_dim0 = 1024 (lo16)
    g1[2] = (unsigned)(32u << 16);                 // tensor_dim1 = 32 (lo16)
    g1[3] = (unsigned)(1024u << 16);               // tile_dim0 = 1024
    g1[4] = 32u | (4u << 16);                      // tile_dim1=32, tile_dim2=4
    g1[5] = 1024u;                                 // tensor_dim0_stride = 1024
    g1[6] = 0u;                                    // dim0_stride hi / dim1_stride lo16
    g1[7] = (unsigned)((unsigned long long)(H * 1024) >> 16); // dim1_stride hi (=16)
    u32x4 g2;
    g2[0] = 4u;                                    // tensor_dim2 = 4
    g2[1] = 0u; g2[2] = 0u; g2[3] = 0u;
    u32x4 g3 = {0u, 0u, 0u, 0u};
    asm volatile("tensor_load_to_lds %0, %1, %2, %3"
                 :: "s"(g0), "s"(g1), "s"(g2), "s"(g3)
                 : "memory");
    __builtin_amdgcn_s_wait_tensorcnt(0);
  }
  __syncthreads();

  // ---- 4-gate WMMA accumulation: A from global (L2-hot h), B from LDS ----
  const __bf16* arow = hbf_r + (size_t)(mt * 16) * H;
  v8f acc[4];
#pragma unroll
  for (int g = 0; g < 4; ++g) acc[g] = v8f{0.f, 0.f, 0.f, 0.f, 0.f, 0.f, 0.f, 0.f};

  for (int k0 = 0; k0 < H; k0 += 32) {
    v16bf ah = load_tile(arow + k0, lane);         // shared across 4 gates
#pragma unroll
    for (int g = 0; g < 4; ++g) {
      const __bf16* brow = lds + (size_t)(g * 32 + h2 * 16) * 1024 + k0;
      v16bf bh = load_tile(brow, lane);            // ds_load path
      acc[g] = __builtin_amdgcn_wmma_f32_16x16x32_bf16(false, ah, false, bh,
                                                       (short)0, acc[g], false, false);
    }
  }

  // ---- fused gate epilogue ----
  int n     = ht * 16 + (lane & 15);
  int mbase = mt * 16 + (lane >> 4) * 8;
  float* hout = out;                     // h lives in d_out[0 .. B*H)
  float* cst  = out + (size_t)B * H;     // c lives in d_out[B*H .. 2*B*H)
#pragma unroll
  for (int v = 0; v < 8; ++v) {
    int m = mbase + v;
    const float* prow = P + (size_t)(t * B + m) * FG;  // bias already folded
    float i = sigmoidf_(acc[0][v] + prow[0 * H + n]);
    float f = sigmoidf_(acc[1][v] + prow[1 * H + n]);
    float g = tanhf(acc[2][v] + prow[2 * H + n]);
    float o = sigmoidf_(acc[3][v] + prow[3 * H + n]);
    float cprev = cst[(size_t)m * H + n];
    float c2 = f * cprev + i * g;
    float h2v = o * tanhf(c2);
    cst[(size_t)m * H + n]   = c2;
    hout[(size_t)m * H + n]  = h2v;
    hbf_w[(size_t)m * H + n] = (__bf16)h2v;   // next step's A operand
  }
}

// ==== fused fallback (small workspace): both GEMMs per step, global loads ====
__global__ __launch_bounds__(256) void lstm_step_fused(
    const __bf16* __restrict__ hbf_r, __bf16* __restrict__ hbf_w,
    const __bf16* __restrict__ whh, const __bf16* __restrict__ xbf,
    const __bf16* __restrict__ wih, const float* __restrict__ bias,
    float* __restrict__ out, int t) {
  int wid  = (blockIdx.x << 3) + (threadIdx.x >> 5);  // 0..255
  int lane = threadIdx.x & 31;
  int mt = wid >> 6;
  int ht = wid & 63;
  const __bf16* arow = hbf_r + (size_t)(mt * 16) * H;
  const __bf16* xrow = xbf + (size_t)(t * B + mt * 16) * E;
  v8f acc[4];
#pragma unroll
  for (int g = 0; g < 4; ++g) acc[g] = v8f{0.f, 0.f, 0.f, 0.f, 0.f, 0.f, 0.f, 0.f};
  for (int k0 = 0; k0 < H; k0 += 32) {
    v16bf ah = load_tile(arow + k0, lane);
    v16bf ax = load_tile(xrow + k0, lane);
#pragma unroll
    for (int g = 0; g < 4; ++g) {
      v16bf bh = load_tile(whh + (size_t)(g * H + ht * 16) * H + k0, lane);
      acc[g] = __builtin_amdgcn_wmma_f32_16x16x32_bf16(false, ah, false, bh,
                                                       (short)0, acc[g], false, false);
      v16bf bx = load_tile(wih + (size_t)(g * H + ht * 16) * E + k0, lane);
      acc[g] = __builtin_amdgcn_wmma_f32_16x16x32_bf16(false, ax, false, bx,
                                                       (short)0, acc[g], false, false);
    }
  }
  int n     = ht * 16 + (lane & 15);
  int mbase = mt * 16 + (lane >> 4) * 8;
  float* hout = out;
  float* cst  = out + (size_t)B * H;
#pragma unroll
  for (int v = 0; v < 8; ++v) {
    int m = mbase + v;
    float i = sigmoidf_(acc[0][v] + bias[0 * H + n]);
    float f = sigmoidf_(acc[1][v] + bias[1 * H + n]);
    float g = tanhf(acc[2][v] + bias[2 * H + n]);
    float o = sigmoidf_(acc[3][v] + bias[3 * H + n]);
    float cprev = cst[(size_t)m * H + n];
    float c2 = f * cprev + i * g;
    float h2v = o * tanhf(c2);
    cst[(size_t)m * H + n]   = c2;
    hout[(size_t)m * H + n]  = h2v;
    hbf_w[(size_t)m * H + n] = (__bf16)h2v;
  }
}

extern "C" void kernel_launch(void* const* d_in, const int* in_sizes, int n_in,
                              void* d_out, int out_size, void* d_ws, size_t ws_size,
                              hipStream_t stream) {
  const float* emb = (const float*)d_in[0];
  const float* Wih = (const float*)d_in[1];
  const float* Whh = (const float*)d_in[2];
  const float* bih = (const float*)d_in[3];
  const float* bhh = (const float*)d_in[4];
  const int*   inp = (const int*)d_in[5];
  float* out = (float*)d_out;

  char* ws = (char*)d_ws;
  __bf16* xbf  = (__bf16*)(ws + XBF_OFF);
  __bf16* wih  = (__bf16*)(ws + WIH_OFF);
  __bf16* whh  = (__bf16*)(ws + WHH_OFF);
  __bf16* hbf  = (__bf16*)(ws + HBF_OFF);   // double-buffered [2][B*H]
  float*  bias = (float*)(ws + BIAS_OFF);
  float*  P    = (float*)(ws + P_OFF);
  const bool use_pre = (ws_size >= P_OFF + P_BYTES);

  init_state<<<(B * H) / 256, 256, 0, stream>>>(out, hbf);
  embed_bf16<<<S * B, 256, 0, stream>>>(emb, inp, xbf);
  cvt_f32_bf16<<<(FG * E / 4) / 256, 256, 0, stream>>>(Wih, wih, FG * E / 4);
  cvt_f32_bf16<<<(FG * H / 4) / 256, 256, 0, stream>>>(Whh, whh, FG * H / 4);
  bias_sum<<<FG / 256, 256, 0, stream>>>(bih, bhh, bias);

  if (use_pre) {
    pregate_gemm<<<(512 * 256) / 8, 256, 0, stream>>>(xbf, wih, bias, P);
    for (int t = 0; t < S; ++t)
      lstm_step_lds<<<32, 256, 4 * 32 * 1024 * 2, stream>>>(
          hbf + (size_t)(t & 1) * B * H, hbf + (size_t)((t + 1) & 1) * B * H,
          whh, P, out, t);
  } else {
    for (int t = 0; t < S; ++t)
      lstm_step_fused<<<32, 256, 0, stream>>>(
          hbf + (size_t)(t & 1) * B * H, hbf + (size_t)((t + 1) & 1) * B * H,
          whh, xbf, wih, bias, out, t);
  }
}
